// GNNClassifier_85564338471311
// MI455X (gfx1250) — matrix operand order repaired and checked by reading, not compiled, not run on hardware
//
#include <hip/hip_runtime.h>
#include <hip/hip_bf16.h>

typedef __attribute__((ext_vector_type(16))) __bf16 v16bf;
typedef __attribute__((ext_vector_type(8)))  __bf16 v8bf;
typedef __attribute__((ext_vector_type(8)))  float  v8f;

// ---------------------------------------------------------------------------
// Utility kernels
// ---------------------------------------------------------------------------
__global__ void zero_f32_kernel(float* __restrict__ p, long long n) {
  long long i = (long long)blockIdx.x * blockDim.x + threadIdx.x;
  if (i < n) p[i] = 0.0f;
}

__global__ void zero_i32_kernel(int* __restrict__ p, int n) {
  int i = blockIdx.x * blockDim.x + threadIdx.x;
  if (i < n) p[i] = 0;
}

__global__ void cvt_f32_bf16_kernel(const float* __restrict__ in,
                                    __bf16* __restrict__ out, long long n) {
  long long i = (long long)blockIdx.x * blockDim.x + threadIdx.x;
  if (i < n) out[i] = (__bf16)in[i];
}

// ---------------------------------------------------------------------------
// Degree / normalization  (deg includes the self loop: deg = indeg + 1)
// ---------------------------------------------------------------------------
__global__ void deg_count_kernel(const int* __restrict__ dst,
                                 int* __restrict__ deg, int E) {
  int e = blockIdx.x * blockDim.x + threadIdx.x;
  if (e < E) atomicAdd(&deg[dst[e]], 1);
}

__global__ void dinv_kernel(const int* __restrict__ deg,
                            float* __restrict__ dinv, int n) {
  int i = blockIdx.x * blockDim.x + threadIdx.x;
  if (i < n) dinv[i] = rsqrtf((float)(deg[i] + 1));
}

// ---------------------------------------------------------------------------
// Pack weights B[K,N] (f32 row-major) into WMMA bf16 fragment order.
// Tile (nt, kt) occupies 512 consecutive bf16 (32 lanes x 16 elems) at offset
// ((nt * (K/32) + kt) * 512). Lane l, elem e holds
//   B[kt*32 + (l>=16 ? 8 : 0) + (e & 7) + (e >= 8 ? 16 : 0)][nt*16 + (l & 15)]
// = the V_WMMA_*_16X16X32_BF16 B-fragment (ISA 7.12.2).
// ---------------------------------------------------------------------------
__global__ void pack_b_kernel(const float* __restrict__ B,
                              __bf16* __restrict__ Bp, int N, int K) {
  long long i = (long long)blockIdx.x * blockDim.x + threadIdx.x;
  long long total = (long long)N * K;
  if (i >= total) return;
  int e    = (int)(i & 15);
  int lane = (int)((i >> 4) & 31);
  long long tile = i >> 9;
  int ksteps = K >> 5;
  int nt = (int)(tile / ksteps);
  int kt = (int)(tile % ksteps);
  int col = nt * 16 + (lane & 15);
  int k   = (kt << 5) + ((lane >> 4) << 3) + (e & 7) + ((e >> 3) << 4);
  Bp[i] = (__bf16)B[(size_t)k * N + col];
}

// ---------------------------------------------------------------------------
// bf16 WMMA GEMM:  C[M,N] = A[M,K] @ Bp   (A bf16 row-major, Bp packed, C f32)
// Each wave computes a 32x64 block: 2 M-tiles x 4 N-tiles (8 accumulators).
// Per K-step: 4 A-frag loads + 4 B-frag loads feed 8 WMMAs.
// M % 32 == 0, N % 64 == 0, K % 32 == 0 guaranteed by caller.
// ---------------------------------------------------------------------------
__global__ void wmma_gemm_bf16_kernel(const __bf16* __restrict__ A,
                                      const __bf16* __restrict__ Bp,
                                      float* __restrict__ C,
                                      int M, int N, int K) {
  const int lane = threadIdx.x & 31;
  const int wave = blockIdx.x * (blockDim.x >> 5) + (threadIdx.x >> 5);
  const int n64  = N >> 6;
  const int total = (M >> 5) * n64;
  if (wave >= total) return;              // wave-uniform: EXEC stays all-ones

  const int mg  = wave / n64;             // 32-row block
  const int ng  = wave % n64;             // group of 4 consecutive n-tiles
  const int half = lane >> 4;
  const int l15  = lane & 15;
  const int ksteps = K >> 5;

  // A fragments: lanes 0-15 take K = {0..7,16..23}, lanes 16-31 {8..15,24..31}
  const __bf16* Arow0 = A + (size_t)(mg * 32 + l15) * K + (half ? 8 : 0);
  const __bf16* Arow1 = Arow0 + (size_t)16 * K;

  // Packed B: base of tile (ng*4 + t, kt) plus this lane's 16 elements.
  const size_t tstride = (size_t)ksteps * 512;   // bf16 elems per n-tile column
  const __bf16* Bbase = Bp + (size_t)(ng * 4) * tstride + lane * 16;

  v8f acc00 = {}, acc01 = {}, acc02 = {}, acc03 = {};
  v8f acc10 = {}, acc11 = {}, acc12 = {}, acc13 = {};
  for (int kt = 0; kt < ksteps; ++kt) {
    const int ko = kt << 5;
    v8bf a00 = *reinterpret_cast<const v8bf*>(Arow0 + ko);
    v8bf a01 = *reinterpret_cast<const v8bf*>(Arow0 + ko + 16);
    v8bf a10 = *reinterpret_cast<const v8bf*>(Arow1 + ko);
    v8bf a11 = *reinterpret_cast<const v8bf*>(Arow1 + ko + 16);
    v16bf af0 = __builtin_shufflevector(a00, a01,
        0, 1, 2, 3, 4, 5, 6, 7, 8, 9, 10, 11, 12, 13, 14, 15);
    v16bf af1 = __builtin_shufflevector(a10, a11,
        0, 1, 2, 3, 4, 5, 6, 7, 8, 9, 10, 11, 12, 13, 14, 15);

    const __bf16* bk = Bbase + (size_t)kt * 512;
    v16bf b0 = *reinterpret_cast<const v16bf*>(bk);
    v16bf b1 = *reinterpret_cast<const v16bf*>(bk + tstride);
    v16bf b2 = *reinterpret_cast<const v16bf*>(bk + 2 * tstride);
    v16bf b3 = *reinterpret_cast<const v16bf*>(bk + 3 * tstride);

    acc00 = __builtin_amdgcn_wmma_f32_16x16x32_bf16(false, af0, false, b0,
                                                    (short)0, acc00, false, false);
    acc01 = __builtin_amdgcn_wmma_f32_16x16x32_bf16(false, af0, false, b1,
                                                    (short)0, acc01, false, false);
    acc02 = __builtin_amdgcn_wmma_f32_16x16x32_bf16(false, af0, false, b2,
                                                    (short)0, acc02, false, false);
    acc03 = __builtin_amdgcn_wmma_f32_16x16x32_bf16(false, af0, false, b3,
                                                    (short)0, acc03, false, false);
    acc10 = __builtin_amdgcn_wmma_f32_16x16x32_bf16(false, af1, false, b0,
                                                    (short)0, acc10, false, false);
    acc11 = __builtin_amdgcn_wmma_f32_16x16x32_bf16(false, af1, false, b1,
                                                    (short)0, acc11, false, false);
    acc12 = __builtin_amdgcn_wmma_f32_16x16x32_bf16(false, af1, false, b2,
                                                    (short)0, acc12, false, false);
    acc13 = __builtin_amdgcn_wmma_f32_16x16x32_bf16(false, af1, false, b3,
                                                    (short)0, acc13, false, false);
  }

  // C fragment store: lanes 0-15 -> rows 0..7, lanes 16-31 -> rows 8..15
  float* Ct0 = C + (size_t)(mg * 32) * N + ng * 64;
  float* Ct1 = Ct0 + (size_t)16 * N;
  const int mofs = half ? 8 : 0;
#pragma unroll
  for (int v = 0; v < 8; ++v) {
    float* r0 = Ct0 + (size_t)(mofs + v) * N + l15;
    r0[0]  = acc00[v];
    r0[16] = acc01[v];
    r0[32] = acc02[v];
    r0[48] = acc03[v];
    float* r1 = Ct1 + (size_t)(mofs + v) * N + l15;
    r1[0]  = acc10[v];
    r1[16] = acc11[v];
    r1[32] = acc12[v];
    r1[48] = acc13[v];
  }
}

// ---------------------------------------------------------------------------
// Edge aggregation: AGG[dst] += H[src] * dinv[src]*dinv[dst]
// grid.x over edges, grid.y over feature chunks of 4 floats.
// ---------------------------------------------------------------------------
__global__ void edge_agg_kernel(const float* __restrict__ H,
                                const int* __restrict__ src,
                                const int* __restrict__ dst,
                                const float* __restrict__ dinv,
                                float* __restrict__ AGG, int E, int F) {
  int e = blockIdx.x * blockDim.x + threadIdx.x;
  if (e >= E) return;
  int c = blockIdx.y << 2;
  int s = src[e], d = dst[e];
  float nrm = dinv[s] * dinv[d];
  const float4 hv = *reinterpret_cast<const float4*>(H + (size_t)s * F + c);
  float* o = AGG + (size_t)d * F + c;
  unsafeAtomicAdd(o + 0, hv.x * nrm);
  unsafeAtomicAdd(o + 1, hv.y * nrm);
  unsafeAtomicAdd(o + 2, hv.z * nrm);
  unsafeAtomicAdd(o + 3, hv.w * nrm);
}

// ---------------------------------------------------------------------------
// GCN epilogue: out = relu(AGG + H*dinv^2 (self loop) + bias), repack to bf16
// ---------------------------------------------------------------------------
__global__ void gcn_finalize_kernel(const float* __restrict__ AGG,
                                    const float* __restrict__ H,
                                    const float* __restrict__ dinv,
                                    const float* __restrict__ bias,
                                    __bf16* __restrict__ Xb,
                                    long long total, int fmask, int fshift) {
  long long i = (long long)blockIdx.x * blockDim.x + threadIdx.x;
  if (i >= total) return;
  int f = (int)(i & fmask);
  int node = (int)(i >> fshift);
  float di = dinv[node];
  float v = AGG[i] + H[i] * di * di + bias[f];
  v = fmaxf(v, 0.0f);
  Xb[i] = (__bf16)v;
}

// Dense epilogue: outF = relu(H + bias)  (kept in f32 for the tiny head GEMM)
__global__ void bias_relu_kernel(const float* __restrict__ H,
                                 const float* __restrict__ bias,
                                 float* __restrict__ outF,
                                 long long total, int fmask) {
  long long i = (long long)blockIdx.x * blockDim.x + threadIdx.x;
  if (i >= total) return;
  float v = H[i] + bias[(int)(i & fmask)];
  outF[i] = fmaxf(v, 0.0f);
}

// Head: out[i, 0..2] = H4[i, :] @ Wc[512,3] + bc   (one node per thread)
__global__ void head_gemm_kernel(const float* __restrict__ H4,
                                 const float* __restrict__ Wc,
                                 const float* __restrict__ bc,
                                 float* __restrict__ out, int n, int K) {
  int i = blockIdx.x * blockDim.x + threadIdx.x;
  if (i >= n) return;
  const float* h = H4 + (size_t)i * K;
  float s0 = bc[0], s1 = bc[1], s2 = bc[2];
  for (int k = 0; k < K; ++k) {
    float hv = h[k];
    s0 += hv * Wc[k * 3 + 0];
    s1 += hv * Wc[k * 3 + 1];
    s2 += hv * Wc[k * 3 + 2];
  }
  out[(size_t)i * 3 + 0] = s0;
  out[(size_t)i * 3 + 1] = s1;
  out[(size_t)i * 3 + 2] = s2;
}

// ---------------------------------------------------------------------------
// Host-side orchestration
// ---------------------------------------------------------------------------
static inline int cdiv(long long a, int b) { return (int)((a + b - 1) / b); }

static void launch_gemm(const __bf16* A, const __bf16* Bp, float* C,
                        int M, int N, int K, hipStream_t s) {
  int tiles = (M >> 5) * (N >> 6);        // one 32x64 block per wave
  int blocks = (tiles + 7) / 8;           // 8 waves (256 threads) per block
  wmma_gemm_bf16_kernel<<<blocks, 256, 0, s>>>(A, Bp, C, M, N, K);
}

extern "C" void kernel_launch(void* const* d_in, const int* in_sizes, int n_in,
                              void* d_out, int out_size, void* d_ws, size_t ws_size,
                              hipStream_t stream) {
  const float* x  = (const float*)d_in[0];
  const int*   ei = (const int*)d_in[1];
  const float* W1 = (const float*)d_in[2];  const float* b1 = (const float*)d_in[3];
  const float* W2 = (const float*)d_in[4];  const float* b2 = (const float*)d_in[5];
  const float* W3 = (const float*)d_in[6];  const float* b3 = (const float*)d_in[7];
  const float* Wi = (const float*)d_in[8];  const float* bi = (const float*)d_in[9];
  const float* Wc = (const float*)d_in[10]; const float* bc = (const float*)d_in[11];
  float* out = (float*)d_out;

  const int IN_DIM = 128, INT_DIM = 512;
  const int N = in_sizes[0] / IN_DIM;       // 100000
  const int E = in_sizes[1] / 2;            // 3200000
  const int* src = ei;
  const int* dst = ei + E;

  // Workspace layout (256B aligned)
  char* ws = (char*)d_ws;
  size_t off = 0;
  auto alloc = [&](size_t bytes) -> void* {
    void* p = ws + off;
    off += (bytes + 255) & ~(size_t)255;
    return p;
  };
  int*    deg  = (int*)alloc((size_t)N * 4);
  float*  dinv = (float*)alloc((size_t)N * 4);
  __bf16* Xb   = (__bf16*)alloc((size_t)N * INT_DIM * 2);   // bf16 activations
  float*  H    = (float*)alloc((size_t)N * INT_DIM * 4);    // GEMM output
  float*  AGG  = (float*)alloc((size_t)N * INT_DIM * 4);    // aggregation accum
  __bf16* Bp   = (__bf16*)alloc((size_t)INT_DIM * INT_DIM * 2); // packed weights

  // ---- degrees & symmetric normalization --------------------------------
  zero_i32_kernel<<<cdiv(N, 256), 256, 0, stream>>>(deg, N);
  deg_count_kernel<<<cdiv(E, 256), 256, 0, stream>>>(dst, deg, E);
  dinv_kernel<<<cdiv(N, 256), 256, 0, stream>>>(deg, dinv, N);

  // ---- input features -> bf16 -------------------------------------------
  long long nx = (long long)N * IN_DIM;
  cvt_f32_bf16_kernel<<<cdiv(nx, 256), 256, 0, stream>>>(x, Xb, nx);

  // ---- three GCN layers ---------------------------------------------------
  struct { const float* W; const float* b; int fin; int fout; } layers[3] = {
    { W1, b1, IN_DIM, IN_DIM }, { W2, b2, IN_DIM, IN_DIM },
    { W3, b3, IN_DIM, INT_DIM } };

  for (int L = 0; L < 3; ++L) {
    int fin = layers[L].fin, fout = layers[L].fout;
    long long nw = (long long)fin * fout;
    pack_b_kernel<<<cdiv(nw, 256), 256, 0, stream>>>(layers[L].W, Bp, fout, fin);

    launch_gemm(Xb, Bp, H, N, fout, fin, stream);        // H = X @ W

    long long tot = (long long)N * fout;
    zero_f32_kernel<<<cdiv(tot, 256), 256, 0, stream>>>(AGG, tot);

    dim3 agrid(cdiv(E, 256), fout >> 2);
    edge_agg_kernel<<<agrid, 256, 0, stream>>>(H, src, dst, dinv, AGG, E, fout);

    int fshift = (fout == 128) ? 7 : 9;
    gcn_finalize_kernel<<<cdiv(tot, 256), 256, 0, stream>>>(
        AGG, H, dinv, layers[L].b, Xb, tot, fout - 1, fshift);
  }

  // ---- dense 512 -> 512 + ReLU -------------------------------------------
  long long nwi = (long long)INT_DIM * INT_DIM;
  pack_b_kernel<<<cdiv(nwi, 256), 256, 0, stream>>>(Wi, Bp, INT_DIM, INT_DIM);
  launch_gemm(Xb, Bp, H, N, INT_DIM, INT_DIM, stream);
  long long tot = (long long)N * INT_DIM;
  bias_relu_kernel<<<cdiv(tot, 256), 256, 0, stream>>>(H, bi, AGG, tot, INT_DIM - 1);

  // ---- head 512 -> 3 ------------------------------------------------------
  head_gemm_kernel<<<cdiv(N, 256), 256, 0, stream>>>(AGG, Wc, bc, out, N, INT_DIM);
}
